// KPConv_73778948211299
// MI455X (gfx1250) — compile-verified, matrix operand order checked by reading, not run on hardware
//
#include <hip/hip_runtime.h>

// ---------------------------------------------------------------------------
// KPConv on gfx1250 via V_WMMA_F32_16X16X32_F16.
//   Prep 1: x (f32) -> xh (packed f16) so gathers are 8-half b128 loads.
//   Prep 2: weights packed into exact per-lane WMMA B-fragments (f16).
//   Stage A (per point): weighted[16k x 64c] = W(16x32h) x Xg(32h x 64c)
//       - lane <-> neighbor h; 8x global_load_b128 gather of own row,
//         h<->c transpose through padded LDS, 4 WMMA per point.
//   Stage B (per 16-pt tile): out[16 x 96] = Y[16 x 960] x Wf[960 x 96]
//       - 30 K-steps x 6 Cout tiles of WMMA; A from padded LDS (conflict-free
//         quads), B streamed as contiguous 32B/lane from L2-resident pack.
// Roofline: ~12.5 GFLOP vs ~40 MB resident in 192MB L2 -> matrix-ALU bound.
// ---------------------------------------------------------------------------

typedef __attribute__((ext_vector_type(16))) _Float16     v16h;
typedef __attribute__((ext_vector_type(8)))  float        v8f;
typedef __attribute__((ext_vector_type(4)))  unsigned int v4u;
typedef __attribute__((ext_vector_type(4)))  float        v4f;
typedef __attribute__((ext_vector_type(4)))  _Float16     v4h;

union Frag  { v16h h; v4u u[2]; };
union Pack8 { v4u u; _Float16 h[8]; };

#define HN          32              // neighbors
#define KPN         15              // kernel points
#define CIN         64
#define COUT        96
#define KP_INV      (1.0f / 0.03f)
#define PTS_PER_WG  16
#define TSTEPS      30              // 960/32 K-steps (k-major kc, skips k=15 pad)
#define NTILES      6               // 96/16 Cout tiles

#define WS_PITCH    40              // halves per k-row   (20 dwords: bank quads)
#define XC_PITCH    56              // halves per c-row   (28 dwords: bank quads)
#define YS_PITCH    1032            // halves per pt-row  (516 dwords == 4 mod 64)
#define SCR_HALVES  (32 * XC_PITCH) // 1792 halves = 3584 B per-wave scratch

// ---------------------------------------------------------------------------
// Prep 1: f32 -> f16 copy of features (coalesced).
// ---------------------------------------------------------------------------
__global__ void cvt_x(const float* __restrict__ x, _Float16* __restrict__ xh, int total4)
{
    int i = blockIdx.x * blockDim.x + threadIdx.x;
    if (i < total4) {
        v4f v = ((const v4f*)x)[i];
        v4h h;
        h.x = (_Float16)v.x; h.y = (_Float16)v.y;
        h.z = (_Float16)v.z; h.w = (_Float16)v.w;
        ((v4h*)xh)[i] = h;
    }
}

// ---------------------------------------------------------------------------
// Prep 2: pack weights[k][c][o] into per-lane WMMA B fragments (f16).
// Fragment f = t*6+nt covers kc in [t*32, t*32+32) (kc = k*64+c), o-tile nt.
// Lane L<16: col o=nt*16+L, rows kc = t*32 + {0..7,16..23}; L>=16: {8..15,24..31}.
// 16 halves per lane stored contiguously (32 B) at (f*32 + L)*32 B.
// ---------------------------------------------------------------------------
__global__ void pack_weights(const float* __restrict__ w, _Float16* __restrict__ wfp)
{
    const int f    = blockIdx.x;          // 0..179
    const int t    = f / NTILES;
    const int nt   = f % NTILES;
    const int lane = threadIdx.x;         // 0..31
    const int o    = nt * 16 + (lane & 15);
    const int kcb  = t * 32 + ((lane < 16) ? 0 : 8);

    Frag v;
#pragma unroll
    for (int i = 0; i < 16; ++i) {
        int kc = kcb + i + ((i >= 8) ? 8 : 0);
        int k  = kc >> 6;
        int c  = kc & 63;
        float val = (k < KPN) ? w[(size_t)(k * CIN + c) * COUT + o] : 0.0f;
        v.h[i] = (_Float16)val;
    }
    v4u* gp = (v4u*)(wfp + ((size_t)f * 32 + lane) * 16);
    gp[0] = v.u[0];
    gp[1] = v.u[1];
}

// ---------------------------------------------------------------------------
// Main kernel: one workgroup = 16 points, 8 waves (wave32), 61.7 KB LDS.
// ---------------------------------------------------------------------------
__global__ __launch_bounds__(256) void kpconv_main(
    const float* __restrict__ q, const float* __restrict__ s,
    const _Float16* __restrict__ xh, const float* __restrict__ kp,
    const int* __restrict__ inds, const _Float16* __restrict__ wfp,
    float* __restrict__ out, int N)
{
    __shared__ __align__(16) _Float16 Scr[8][SCR_HALVES];        // 28 KB
    __shared__ __align__(16) _Float16 Ys[PTS_PER_WG][YS_PITCH];  // 32.25 KB

    const int tid  = threadIdx.x;
    const int wave = tid >> 5;
    const int lane = tid & 31;
    const int base = blockIdx.x * PTS_PER_WG;
    const int h0   = (lane < 16) ? 0 : 8;   // fragment half-row base
    const int l16  = lane & 15;
    const int mhi  = (lane < 16) ? 0 : 8;
    _Float16* scr  = Scr[wave];             // per-wave transpose scratch

    // ---------------- Stage A: each wave owns 2 points --------------------
    for (int sp = 0; sp < 2; ++sp) {
        const int p = wave * 2 + sp;
        const int n = base + p;
        if (n < N) {
            // 1) influence weights: lane <-> neighbor h; store k x h to scratch
            const int idx = inds[(size_t)n * HN + lane];
            float sx, sy, sz;
            if (idx < N) { sx = s[idx*3+0]; sy = s[idx*3+1]; sz = s[idx*3+2]; }
            else         { sx = sy = sz = 1.0e6f; }          // shadow point
            const float rx = sx - q[n*3+0];
            const float ry = sy - q[n*3+1];
            const float rz = sz - q[n*3+2];
#pragma unroll
            for (int k = 0; k < KPN; ++k) {
                float dx = rx - kp[k*3+0];
                float dy = ry - kp[k*3+1];
                float dz = rz - kp[k*3+2];
                float wv = fmaxf(1.0f - sqrtf(dx*dx + dy*dy + dz*dz) * KP_INV, 0.0f);
                scr[k * WS_PITCH + lane] = (_Float16)wv;
            }
            scr[15 * WS_PITCH + lane] = (_Float16)0.0f;

            // 2) A fragment = influence matrix [16k x 32h] (LDS in-order: safe
            //    to load before the gather below overwrites the scratch)
            Frag a;
            a.u[0] = *(const v4u*)&scr[l16 * WS_PITCH + h0];
            a.u[1] = *(const v4u*)&scr[l16 * WS_PITCH + h0 + 16];

            // 3) gather this lane's neighbor feature row (f16, b128 loads),
            //    transpose h<->c through scratch, WMMA per 16-channel tile
            const _Float16* xrow = xh + (size_t)((idx < N) ? idx : 0) * CIN;
#pragma unroll
            for (int half = 0; half < 2; ++half) {
#pragma unroll
                for (int cq = 0; cq < 4; ++cq) {
                    Pack8 pk;
                    pk.u = *(const v4u*)(xrow + half * 32 + cq * 8);
#pragma unroll
                    for (int m = 0; m < 8; ++m)
                        scr[(cq * 8 + m) * XC_PITCH + lane] = pk.h[m];
                }
#pragma unroll
                for (int jj = 0; jj < 2; ++jj) {
                    const int cl = jj * 16 + l16;        // local channel row
                    Frag b;
                    b.u[0] = *(const v4u*)&scr[cl * XC_PITCH + h0];
                    b.u[1] = *(const v4u*)&scr[cl * XC_PITCH + h0 + 16];
                    v8f acc = {};
                    acc = __builtin_amdgcn_wmma_f32_16x16x32_f16(
                              false, a.h, false, b.h, (short)0, acc, false, false);
                    const int c = half * 32 + jj * 16 + l16;
#pragma unroll
                    for (int r = 0; r < 8; ++r)
                        Ys[p][(r + mhi) * CIN + c] = (_Float16)acc[r];
                }
            }
        } else {
            // keep Ys defined for stage B reads
            for (int kc = lane; kc < YS_PITCH; kc += 32)
                Ys[p][kc] = (_Float16)0.0f;
        }
    }

    __syncthreads();

    // ---------------- Stage B: waves 0..5, one Cout tile each -------------
    if (wave < NTILES) {
        const int nt = wave;
        v8f acc = {};
        for (int t = 0; t < TSTEPS; ++t) {
            Frag a, b;
            // A fragment: Y rows = points; padded pitch -> disjoint bank quads
            a.u[0] = *(const v4u*)&Ys[l16][t * 32 + h0];
            a.u[1] = *(const v4u*)&Ys[l16][t * 32 + h0 + 16];
            // B fragment: pre-packed weights, contiguous 32 B per lane (L2 hot)
            const v4u* gp = (const v4u*)(wfp +
                ((size_t)(t * NTILES + nt) * 32 + lane) * 16);
            b.u[0] = gp[0];
            b.u[1] = gp[1];
            acc = __builtin_amdgcn_wmma_f32_16x16x32_f16(
                      false, a.h, false, b.h, (short)0, acc, false, false);
        }
#pragma unroll
        for (int r = 0; r < 8; ++r) {
            int pp = r + mhi;
            int n  = base + pp;
            if (n < N)
                out[(size_t)n * COUT + nt * 16 + l16] = acc[r];
        }
    }
}

// ---------------------------------------------------------------------------
extern "C" void kernel_launch(void* const* d_in, const int* in_sizes, int n_in,
                              void* d_out, int out_size, void* d_ws, size_t ws_size,
                              hipStream_t stream)
{
    const float* q    = (const float*)d_in[0];   // [N,3]
    const float* s    = (const float*)d_in[1];   // [N,3]
    const float* x    = (const float*)d_in[2];   // [N,64]
    const float* w    = (const float*)d_in[3];   // [15,64,96]
    const float* kp   = (const float*)d_in[4];   // [15,3]
    const int*   inds = (const int*)d_in[5];     // [N,32]
    float*       out  = (float*)d_out;           // [N,96]
    const int N = in_sizes[0] / 3;

    // Workspace layout: [0, 184320) packed weight frags; then f16 features.
    const size_t wfBytes = (size_t)(TSTEPS * NTILES) * 32 * 32;   // 184320
    const size_t xhOff   = (wfBytes + 255) & ~(size_t)255;
    const size_t need    = xhOff + (size_t)N * CIN * sizeof(_Float16);
    if (ws_size < need) return;                                    // scratch too small

    _Float16* wfp = (_Float16*)d_ws;
    _Float16* xhp = (_Float16*)((char*)d_ws + xhOff);

    const int total4 = N * CIN / 4;
    cvt_x<<<(total4 + 255) / 256, 256, 0, stream>>>(x, xhp, total4);
    pack_weights<<<TSTEPS * NTILES, 32, 0, stream>>>(w, wfp);

    const int nwg = (N + PTS_PER_WG - 1) / PTS_PER_WG;
    kpconv_main<<<nwg, 256, 0, stream>>>(q, s, xhp, kp, inds, wfp, out, N);
}
